// MultiHeadSelfAttention_65712999629007
// MI455X (gfx1250) — matrix operand order repaired and checked by reading, not compiled, not run on hardware
//
#include <hip/hip_runtime.h>
#include <hip/hip_bf16.h>

typedef _Float16 f16;
typedef __attribute__((ext_vector_type(4)))  float vf4;
typedef __attribute__((ext_vector_type(8)))  float vf8;
typedef __attribute__((ext_vector_type(4)))  _Float16 vh4;
typedef __attribute__((ext_vector_type(8)))  _Float16 vh8;
typedef __attribute__((ext_vector_type(16))) _Float16 vh16;

#define EMB 768
#define SEQ 4096
#define NHEAD 12
#define DH 64

// ---------------- WMMA fragment helpers (wave32, 16x16x32 f16) ----------------
// A-matrix 16x32 f16 layout: lane<16 -> row=lane, halves K[0..7] & K[16..23];
// lane>=16 -> row=lane-16, halves K[8..15] & K[24..31].  Caller pre-offsets p by 0/8.
__device__ __forceinline__ vh16 frag_a(const f16* p) {
  vh8 lo = *(const vh8*)p;
  vh8 hi = *(const vh8*)(p + 16);
  vh16 r;
#pragma unroll
  for (int i = 0; i < 8; ++i) { r[i] = lo[i]; r[i + 8] = hi[i]; }
  return r;
}
// B-matrix 32x16 f16 layout: lane<16 -> col=lane, halves K[0..7] & K[16..23];
// lane>=16 -> col=lane-16, halves K[8..15] & K[24..31].  Caller pre-offsets p by 0/8.
__device__ __forceinline__ vh16 frag_b(const f16* p) {
  vh8 lo = *(const vh8*)p;
  vh8 hi = *(const vh8*)(p + 16);
  vh16 r;
#pragma unroll
  for (int i = 0; i < 8; ++i) { r[i] = lo[i]; r[i + 8] = hi[i]; }
  return r;
}
__device__ __forceinline__ vf8 wmma16x16x32(vh16 a, vh16 b, vf8 c) {
  return __builtin_amdgcn_wmma_f32_16x16x32_f16(false, a, false, b, (short)0, c,
                                                false, false);
}

// ---------------- CDNA5 async global->LDS copy (ASYNCcnt-tracked) ------------
// Each lane copies 16B: LDS[lds_off] = MEM[gaddr].  Generic LDS pointers carry
// the LDS byte offset in their low 32 bits (flat aperture encoding).
__device__ __forceinline__ void async_copy_b128(const void* lds_dst,
                                                const void* gsrc) {
  unsigned lds_off = (unsigned)(uintptr_t)lds_dst;
  unsigned long long ga = (unsigned long long)(uintptr_t)gsrc;
  asm volatile("global_load_async_to_lds_b128 %0, %1, off"
               :: "v"(lds_off), "v"(ga)
               : "memory");
}
__device__ __forceinline__ void wait_async0() {
  asm volatile("s_wait_asynccnt 0" ::: "memory");
}

// ---------------- GEMM: C[M,N] = A[M,K] @ W[K,N] + bias ----------------
// Workgroup tile 128x64 (8 waves x 16 rows), K-step 32.
#define MT 128
#define NT 64
#define KT 32
#define LDA 40  // halves (80B, multiple of 16B)
#define LDB 40

template <bool A_F16, bool OUT_F16>
__global__ __launch_bounds__(256) void gemm_bias_wmma(
    const void* __restrict__ Ain, const float* __restrict__ W,
    const float* __restrict__ bias, void* __restrict__ Cout,
    int M, int N, int K) {
  __shared__ alignas(16) f16 sA[MT * LDA];
  __shared__ alignas(16) f16 sB[NT * LDB];  // W tile, transposed: sB[n][k]

  const int t  = threadIdx.x;
  const int wv = t >> 5, ln = t & 31;
  const int m0 = blockIdx.y * MT, n0 = blockIdx.x * NT;
  const int koffA = (ln >> 4) ? 8 : 0;
  const int koffB = (ln >> 4) ? 8 : 0;

  vf8 acc[4] = {};

  for (int k0 = 0; k0 < K; k0 += KT) {
    // ---- stage A tile 128x32 as f16 ----
    if constexpr (A_F16) {
      // Already f16: pure async DMA global->LDS, no VGPR round-trip.
      const f16* Af = (const f16*)Ain;
#pragma unroll
      for (int i = 0; i < 2; ++i) {
        int idx = t + i * 256;               // b128 units, 512 total
        int row = idx >> 2, cv = idx & 3;
        async_copy_b128(sA + row * LDA + cv * 8,
                        Af + (size_t)(m0 + row) * K + k0 + cv * 8);
      }
    } else {
      const float* Af = (const float*)Ain;
#pragma unroll
      for (int i = 0; i < 4; ++i) {
        int idx = t + i * 256;               // vf4 units, 1024 total
        int row = idx >> 3, cv = idx & 7;
        vf4 x = *(const vf4*)(Af + (size_t)(m0 + row) * K + k0 + cv * 4);
        vh4 hx;
#pragma unroll
        for (int j = 0; j < 4; ++j) hx[j] = (f16)x[j];
        *(vh4*)(sA + row * LDA + cv * 4) = hx;
      }
    }
    // ---- stage W tile 32x64, transposed to sB[n][k] (f32 -> f16 convert) ----
#pragma unroll
    for (int i = 0; i < 2; ++i) {
      int idx = t + i * 256;                 // vf4 units, 512 total
      int kk = idx >> 4, nv = idx & 15;
      vf4 x = *(const vf4*)(W + (size_t)(k0 + kk) * N + n0 + nv * 4);
#pragma unroll
      for (int j = 0; j < 4; ++j) sB[(nv * 4 + j) * LDB + kk] = (f16)x[j];
    }
    if constexpr (A_F16) wait_async0();
    __syncthreads();

    const f16* pa = sA + (wv * 16 + (ln & 15)) * LDA + koffA;
    vh16 a = frag_a(pa);
#pragma unroll
    for (int j = 0; j < 4; ++j) {
      const f16* pb = sB + (j * 16 + (ln & 15)) * LDB + koffB;
      acc[j] = wmma16x16x32(a, frag_b(pb), acc[j]);
    }
    __syncthreads();
  }

  // ---- epilogue: C layout lane<16 -> row=r, lane>=16 -> row=r+8; col=lane&15 ----
#pragma unroll
  for (int j = 0; j < 4; ++j) {
    int col = n0 + j * 16 + (ln & 15);
    float bv = bias[col];
#pragma unroll
    for (int r = 0; r < 8; ++r) {
      int row = m0 + wv * 16 + ((ln < 16) ? r : r + 8);
      float val = acc[j][r] + bv;
      if constexpr (OUT_F16)
        ((f16*)Cout)[(size_t)row * N + col] = (f16)val;
      else
        ((float*)Cout)[(size_t)row * N + col] = val;
    }
  }
}

// ---------------- Flash attention (causal), f16 in / f16 out, f32 accum ------
// 1 WG = (64-query block, head). 4 waves x 16 query rows. Dh = 64.
#define BQ 64
#define BK 64
#define SS 68  // f32 score row stride
#define PS 72  // f16 P row stride (144B)
#define VTS 72 // f16 V^T row stride
#define KS 72  // f16 K tile row stride

__global__ __launch_bounds__(128) void flash_attn_wmma(
    const f16* __restrict__ Q, const f16* __restrict__ Kf,
    const f16* __restrict__ Vf, f16* __restrict__ O) {
  __shared__ alignas(16) float sS[BQ * SS];
  __shared__ alignas(16) f16 sP[BQ * PS];
  __shared__ alignas(16) f16 sVt[DH * VTS];
  __shared__ alignas(16) f16 sK[BK * KS];
  __shared__ float sAlpha[BQ];

  const int t  = threadIdx.x;
  const int wv = t >> 5, ln = t & 31;
  const int h  = blockIdx.y, qb = blockIdx.x;
  const int koffA = (ln >> 4) ? 8 : 0;
  const int koffB = (ln >> 4) ? 8 : 0;
  const float scale = 0.125f;  // 1/sqrt(64)

  // Q fragments for this wave's 16 rows, kept in registers across the kb loop.
  vh16 aq[2];
  {
    int row = qb * BQ + wv * 16 + (ln & 15);
    const f16* qp = Q + (size_t)row * EMB + h * DH;
#pragma unroll
    for (int s = 0; s < 2; ++s) aq[s] = frag_a(qp + s * 32 + koffA);
  }

  vf8 accO[4] = {};
  float m_i = -1e30f, l_i = 0.0f;  // per-row state, owned by threads t<64

  for (int kb = 0; kb <= qb; ++kb) {
    // prefetch next K/V block into cache
    if (kb + 1 <= qb) {
      const f16* nk = Kf + (size_t)((kb + 1) * BK + ln) * EMB + h * DH;
      const f16* nv = Vf + (size_t)((kb + 1) * BK + ln) * EMB + h * DH;
      __builtin_prefetch(nk, 0, 1);
      __builtin_prefetch(nv, 0, 1);
    }
    // ---- async-DMA K block 64x64 f16 into sK[kk][d] (shared by all waves) ----
#pragma unroll
    for (int i = 0; i < 4; ++i) {
      int idx = t + i * 128;                 // b128 units, 512 total
      int kk = idx >> 3, dv = idx & 7;
      async_copy_b128(sK + kk * KS + dv * 8,
                      Kf + (size_t)(kb * BK + kk) * EMB + h * DH + dv * 8);
    }
    // ---- stage V block transposed: sVt[d][kk] (needs lane transpose) ----
#pragma unroll
    for (int i = 0; i < 4; ++i) {
      int idx = t + i * 128;                 // vh8 units, 512 total
      int kk = idx >> 3, dv = idx & 7;
      vh8 x = *(const vh8*)(Vf + (size_t)(kb * BK + kk) * EMB + h * DH + dv * 8);
#pragma unroll
      for (int j = 0; j < 8; ++j) sVt[(dv * 8 + j) * VTS + kk] = x[j];
    }
    wait_async0();
    __syncthreads();

    // ---- S = Q @ K^T (B-fragments from the LDS-staged K tile) ----
    vf8 accS[4] = {};
#pragma unroll
    for (int s = 0; s < 2; ++s) {
#pragma unroll
      for (int j = 0; j < 4; ++j) {
        const f16* kp = sK + (j * 16 + (ln & 15)) * KS + s * 32 + koffB;
        accS[j] = wmma16x16x32(aq[s], frag_b(kp), accS[j]);
      }
    }
    // ---- scale + causal mask in registers, write to LDS ----
#pragma unroll
    for (int j = 0; j < 4; ++j) {
#pragma unroll
      for (int r = 0; r < 8; ++r) {
        int rl = wv * 16 + ((ln < 16) ? r : r + 8);
        int cl = j * 16 + (ln & 15);
        float val = accS[j][r] * scale;
        if (kb == qb && cl > rl) val = -1e30f;
        sS[rl * SS + cl] = val;
      }
    }
    __syncthreads();
    // ---- online softmax: one row per thread (t<64) ----
    if (t < BQ) {
      float vmax = -1e30f;
      for (int c = 0; c < BK; ++c) vmax = fmaxf(vmax, sS[t * SS + c]);
      float m_new = fmaxf(m_i, vmax);
      float alpha = __expf(m_i - m_new);
      float sum = 0.0f;
      for (int c = 0; c < BK; ++c) {
        float p = __expf(sS[t * SS + c] - m_new);
        sum += p;
        sP[t * PS + c] = (f16)p;
      }
      l_i = l_i * alpha + sum;
      m_i = m_new;
      sAlpha[t] = alpha;
    }
    __syncthreads();
    // ---- rescale O, then O += P @ V ----
#pragma unroll
    for (int j = 0; j < 4; ++j) {
#pragma unroll
      for (int r = 0; r < 8; ++r) {
        int rl = wv * 16 + ((ln < 16) ? r : r + 8);
        accO[j][r] *= sAlpha[rl];
      }
    }
#pragma unroll
    for (int s = 0; s < 2; ++s) {
      const f16* pp = sP + (wv * 16 + (ln & 15)) * PS + s * 32 + koffA;
      vh16 ap = frag_a(pp);
#pragma unroll
      for (int j = 0; j < 4; ++j) {
        const f16* vp = sVt + (j * 16 + (ln & 15)) * VTS + s * 32 + koffB;
        accO[j] = wmma16x16x32(ap, frag_b(vp), accO[j]);
      }
    }
    __syncthreads();
  }

  // ---- finalize: O /= l, write concat layout [S][E] as f16 ----
  if (t < BQ) sAlpha[t] = 1.0f / l_i;
  __syncthreads();
#pragma unroll
  for (int j = 0; j < 4; ++j) {
#pragma unroll
    for (int r = 0; r < 8; ++r) {
      int rl = wv * 16 + ((ln < 16) ? r : r + 8);
      int row = qb * BQ + rl;
      int col = h * DH + j * 16 + (ln & 15);
      O[(size_t)row * EMB + col] = (f16)(accO[j][r] * sAlpha[rl]);
    }
  }
}

// ---------------------------------------------------------------------------
extern "C" void kernel_launch(void* const* d_in, const int* in_sizes, int n_in,
                              void* d_out, int out_size, void* d_ws,
                              size_t ws_size, hipStream_t stream) {
  (void)in_sizes; (void)n_in; (void)out_size; (void)ws_size;
  const float* q  = (const float*)d_in[0];
  const float* k  = (const float*)d_in[1];
  const float* v  = (const float*)d_in[2];
  // d_in[3] = mask: causal, computed analytically in the kernel; never read.
  const float* wq = (const float*)d_in[4];
  const float* bq = (const float*)d_in[5];
  const float* wk = (const float*)d_in[6];
  const float* bk = (const float*)d_in[7];
  const float* wv = (const float*)d_in[8];
  const float* bv = (const float*)d_in[9];
  const float* wo = (const float*)d_in[10];
  const float* bo = (const float*)d_in[11];

  const size_t BUF = (size_t)SEQ * EMB * sizeof(f16);  // 6.29 MB each
  f16* Qf = (f16*)d_ws;
  f16* Kf = (f16*)((char*)d_ws + BUF);
  f16* Vf = (f16*)((char*)d_ws + 2 * BUF);
  f16* Of = (f16*)((char*)d_ws + 3 * BUF);

  dim3 gg(EMB / NT, SEQ / MT);  // (12, 32)
  gemm_bias_wmma<false, true><<<gg, 256, 0, stream>>>(q, wq, bq, Qf, SEQ, EMB, EMB);
  gemm_bias_wmma<false, true><<<gg, 256, 0, stream>>>(k, wk, bk, Kf, SEQ, EMB, EMB);
  gemm_bias_wmma<false, true><<<gg, 256, 0, stream>>>(v, wv, bv, Vf, SEQ, EMB, EMB);

  flash_attn_wmma<<<dim3(SEQ / BQ, NHEAD), 128, 0, stream>>>(Qf, Kf, Vf, Of);

  gemm_bias_wmma<true, false><<<gg, 256, 0, stream>>>(Of, wo, bo, d_out, SEQ, EMB, EMB);
}
